// MegaBlock_30296699306420
// MI455X (gfx1250) — matrix operand order repaired and checked by reading, not compiled, not run on hardware
//
#include <hip/hip_runtime.h>

// ---------------- problem constants ----------------
constexpr int Bc = 8, Lc = 2048, Dc = 1024, Zc = 128, Vc = 2048, Nc = 16;
constexpr int Mt = Bc * Lc;                 // 16384 total rows
constexpr int COLS = Zc + Vc + 2 * Dc;      // 4224 (mx_w output cols)

// ---------------- WMMA vector types ----------------
typedef __attribute__((ext_vector_type(16))) __bf16 v16bf;
typedef __attribute__((ext_vector_type(8)))  float  v8f;

// ---------------- bf16 helpers (RNE) ----------------
__device__ __forceinline__ unsigned short f2bf(float f) {
    unsigned int u = __builtin_bit_cast(unsigned int, f);
    unsigned int r = u + 0x7FFFu + ((u >> 16) & 1u);
    return (unsigned short)(r >> 16);
}
__device__ __forceinline__ float bf2f(unsigned short h) {
    unsigned int u = ((unsigned int)h) << 16;
    return __builtin_bit_cast(float, u);
}
__device__ __forceinline__ float silu_f(float x) { return x / (1.f + __expf(-x)); }
__device__ __forceinline__ float sigm_f(float x) { return 1.f / (1.f + __expf(-x)); }

// ---------------- CDNA5 async global -> LDS copy (16B per lane) ----------------
__device__ __forceinline__ void async_b128_to_lds(void* lds, const void* g) {
    unsigned int loff = (unsigned int)reinterpret_cast<unsigned long long>(lds);
    asm volatile("global_load_async_to_lds_b128 %0, %1, off"
                 :: "v"(loff), "v"(g) : "memory");
}

// ============================================================
// LayerNorm: one block per row of D=1024. Writes f32 and/or bf16.
// ============================================================
__global__ __launch_bounds__(256) void ln_kernel(const float* __restrict__ X,
                                                 const float* __restrict__ w,
                                                 const float* __restrict__ b,
                                                 float* __restrict__ Yf,
                                                 unsigned short* __restrict__ Yb) {
    __shared__ float s1[256], s2[256];
    const int row = blockIdx.x;
    const float* x = X + (size_t)row * Dc;
    float v[4], sum = 0.f, sq = 0.f;
#pragma unroll
    for (int i = 0; i < 4; i++) {
        float t = x[threadIdx.x + i * 256];
        v[i] = t; sum += t; sq += t * t;
    }
    s1[threadIdx.x] = sum; s2[threadIdx.x] = sq; __syncthreads();
    for (int st = 128; st > 0; st >>= 1) {
        if (threadIdx.x < st) { s1[threadIdx.x] += s1[threadIdx.x + st]; s2[threadIdx.x] += s2[threadIdx.x + st]; }
        __syncthreads();
    }
    const float mu = s1[0] * (1.f / Dc);
    const float var = s2[0] * (1.f / Dc) - mu * mu;
    const float rs = rsqrtf(var + 1e-5f);
#pragma unroll
    for (int i = 0; i < 4; i++) {
        int c = threadIdx.x + i * 256;
        float y = (v[i] - mu) * rs * w[c] + b[c];
        if (Yf) Yf[(size_t)row * Dc + c] = y;
        if (Yb) Yb[(size_t)row * Dc + c] = f2bf(y);
    }
}

// ============================================================
// EMA as exact linear recurrence (replaces FFT conv):
//   s_n[t] = q_n * s_n[t-1] + x[t];  ema[t] = sum_n c_n * s_n[t]
// then mx = silu(ema + xn*omega) stored bf16. One thread per (b,d).
// ============================================================
__global__ __launch_bounds__(256) void ema_mx_kernel(const float* __restrict__ xn,
                                                     const float* __restrict__ delta,
                                                     const float* __restrict__ alpha,
                                                     const float* __restrict__ beta,
                                                     const float* __restrict__ gamma,
                                                     const float* __restrict__ omega,
                                                     unsigned short* __restrict__ mx) {
    const int tid = blockIdx.x * 256 + threadIdx.x;   // 0 .. B*D-1
    const int b = tid / Dc, d = tid % Dc;
    float q[Nc], c[Nc], s[Nc];
#pragma unroll
    for (int n = 0; n < Nc; n++) {
        float p = sigm_f(delta[d * Nc + n]);
        q[n] = 1.f - p * sigm_f(alpha[d * Nc + n]);
        c[n] = p * beta[d * Nc + n] * gamma[d * Nc + n] * 0.25f; // 1/sqrt(16)
        s[n] = 0.f;
    }
    const float om = omega[d];
    const float* xp = xn + (size_t)b * Lc * Dc + d;
    unsigned short* mp = mx + (size_t)b * Lc * Dc + d;
    for (int t = 0; t < Lc; t++) {
        float x = xp[(size_t)t * Dc];
        float e = 0.f;
#pragma unroll
        for (int n = 0; n < Nc; n++) { s[n] = q[n] * s[n] + x; e += c[n] * s[n]; }
        mp[(size_t)t * Dc] = f2bf(silu_f(e + x * om));
    }
}

// ============================================================
// Weight prep: f32 [K][N] -> bf16 [N][K] (B-fragment friendly)
// ============================================================
__global__ __launch_bounds__(256) void wtrans_kernel(const float* __restrict__ W,
                                                     unsigned short* __restrict__ Wt,
                                                     int Kd, int Nd) {
    size_t t = (size_t)blockIdx.x * 256 + threadIdx.x;
    size_t k = t / Nd, n = t % Nd;
    Wt[n * (size_t)Kd + k] = f2bf(W[t]);
}

// ============================================================
// WMMA GEMM with async global->LDS double-buffered staging.
// C[M][N] = act(A[M][K] * Bt[N][K]^T + bias + add)
// Block = 256 thr = 8 waves; tile 64(M) x 128(N); wave = 16x64.
// Per k-step: A tile 64x32 bf16 (4KB), B tile 128x32 bf16 (8KB),
// copied via 3 global_load_async_to_lds_b128 per wave; rows padded
// to 80B so lane-parallel b128 LDS reads hit disjoint bank groups.
// All 5 fragments are loaded to distinct registers before the
// 4-WMMA burst -> single DScnt wait per k-step, back-to-back WMMAs.
// OUTMODE: 0 = f32 [M][N], 1 = bf16 [M][N], 2 = bf16 transposed [N][ldc]
// ACT: 0 none, 1 silu, 2 sigmoid
// ============================================================
template <int ACT, int OUTMODE>
__global__ __launch_bounds__(256) void gemm_wmma(const unsigned short* __restrict__ A,
                                                 const unsigned short* __restrict__ Bt,
                                                 const float* __restrict__ bias,
                                                 const float* __restrict__ add,
                                                 float* __restrict__ Cf,
                                                 unsigned short* __restrict__ Cb,
                                                 int K, int lda, int ldb, int ldc) {
    __shared__ __align__(16) unsigned short la[2][64][40];    // 32 data + 8 pad
    __shared__ __align__(16) unsigned short lb[2][128][40];

    const int t = threadIdx.x;
    const int wave = t >> 5;
    const int lane = t & 31;
    const int mBlock = blockIdx.y * 64;
    const int nBlock = blockIdx.x * 128;

    // copy-job coordinates (16B segments)
    const int ar = t >> 2;            // A row 0..63
    const int as = (t & 3) * 8;       // A seg offset (elements)

    auto issue = [&](int buf, int k0) {
        const unsigned short* ag = A + (size_t)(mBlock + ar) * lda + k0 + as;
        async_b128_to_lds(&la[buf][ar][as], ag);
#pragma unroll
        for (int i = 0; i < 2; i++) {
            int job = t + i * 256;
            int br = job >> 2;        // B row 0..127
            int bs = (job & 3) * 8;
            const unsigned short* bg = Bt + (size_t)(nBlock + br) * ldb + k0 + bs;
            async_b128_to_lds(&lb[buf][br][bs], bg);
        }
    };

    v8f acc[4] = {};
    const int mSub = (wave & 3) * 16;
    const int nSub = (wave >> 2) * 64;

    auto compute = [&](int cur) {
        union { uint4 q[2]; v16bf v; } af, bfr[4];
        // A fragment (16x32): lane<16 -> K {0..7,16..23}, lane>=16 -> +8
        const unsigned short* ap = &la[cur][mSub + (lane & 15)][(lane >> 4) * 8];
        af.q[0] = *(const uint4*)(ap);
        af.q[1] = *(const uint4*)(ap + 16);
#pragma unroll
        for (int j = 0; j < 4; j++) {
            // B fragment (32x16): lane<16 -> K {0..7,16..23}, lane>=16 -> +8
            const unsigned short* bp = &lb[cur][nSub + j * 16 + (lane & 15)][(lane >> 4) * 8];
            bfr[j].q[0] = *(const uint4*)(bp);
            bfr[j].q[1] = *(const uint4*)(bp + 16);
        }
#pragma unroll
        for (int j = 0; j < 4; j++)
            acc[j] = __builtin_amdgcn_wmma_f32_16x16x32_bf16(
                false, af.v, false, bfr[j].v, (short)0, acc[j], false, false);
    };

    const int ksteps = K >> 5;
    issue(0, 0);
    int kt = 0;
    for (; kt < ksteps - 1; kt++) {                      // steady state: always prefetch
        issue((kt + 1) & 1, (kt + 1) << 5);
        asm volatile("s_wait_asynccnt 0x3" ::: "memory"); // current buffer landed
        __syncthreads();
        compute(kt & 1);
        __syncthreads();                                  // safe to overwrite other buffer
    }
    asm volatile("s_wait_asynccnt 0x0" ::: "memory");     // tail
    __syncthreads();
    compute(kt & 1);

    // Epilogue: C/D layout: VGPR r -> row mBase + (lane>>4)*8 + r, col nBase + (lane&15)
    const int mrow = mBlock + mSub + (lane >> 4) * 8;
#pragma unroll
    for (int j = 0; j < 4; j++) {
        const int n = nBlock + nSub + j * 16 + (lane & 15);
        const float bv = bias ? bias[n] : 0.f;
#pragma unroll
        for (int r = 0; r < 8; r++) {
            const int m = mrow + r;
            float val = acc[j][r] + bv;
            if (add) val += add[(size_t)m * ldc + n];
            if (ACT == 1) val = silu_f(val);
            else if (ACT == 2) val = sigm_f(val);
            if (OUTMODE == 0)      Cf[(size_t)m * ldc + n] = val;
            else if (OUTMODE == 1) Cb[(size_t)m * ldc + n] = f2bf(val);
            else                   Cb[(size_t)n * ldc + m] = f2bf(val);
        }
    }
}

// ============================================================
// qh/kh from z:  qh = (z*g0 + b0) * Z^-0.5 ; kh = z*g1 + b1
// ============================================================
__global__ __launch_bounds__(256) void qhkh_kernel(const unsigned short* __restrict__ z,
                                                   const float* __restrict__ g,
                                                   const float* __restrict__ be,
                                                   unsigned short* __restrict__ qh,
                                                   unsigned short* __restrict__ kh) {
    size_t t = (size_t)blockIdx.x * 256 + threadIdx.x;   // over M*Z
    int j = (int)(t % Zc);
    float zv = bf2f(z[t]);
    qh[t] = f2bf((zv * g[j] + be[j]) * 0.08838834764831845f); // 1/sqrt(128)
    kh[t] = f2bf(zv * g[Zc + j] + be[Zc + j]);
}

// ============================================================
// Row softmax with fused relative-position bias; bf16 output.
// bias[l][m] = rel_bias[L-1 + m - l]   (MAXPOS == L)
// ============================================================
__global__ __launch_bounds__(256) void softmax_kernel(const float* __restrict__ S,
                                                      const float* __restrict__ rb,
                                                      unsigned short* __restrict__ A) {
    __shared__ float red[256];
    const int l = blockIdx.x;
    const float* srow = S + (size_t)l * Lc;
    const float* bl = rb + (Lc - 1) - l;
    float v[8], mx = -3.4e38f;
#pragma unroll
    for (int i = 0; i < 8; i++) {
        int m = threadIdx.x + i * 256;
        float t = srow[m] + bl[m];
        v[i] = t; mx = fmaxf(mx, t);
    }
    red[threadIdx.x] = mx; __syncthreads();
    for (int st = 128; st > 0; st >>= 1) {
        if (threadIdx.x < st) red[threadIdx.x] = fmaxf(red[threadIdx.x], red[threadIdx.x + st]);
        __syncthreads();
    }
    mx = red[0]; __syncthreads();
    float sum = 0.f;
#pragma unroll
    for (int i = 0; i < 8; i++) { v[i] = __expf(v[i] - mx); sum += v[i]; }
    red[threadIdx.x] = sum; __syncthreads();
    for (int st = 128; st > 0; st >>= 1) {
        if (threadIdx.x < st) red[threadIdx.x] += red[threadIdx.x + st];
        __syncthreads();
    }
    const float inv = 1.f / red[0];
    unsigned short* arow = A + (size_t)l * Lc;
#pragma unroll
    for (int i = 0; i < 8; i++) {
        int m = threadIdx.x + i * 256;
        arow[m] = f2bf(v[i] * inv);
    }
}

// ============================================================
// hr = bf16(h * r)   (elementwise, per-batch slab)
// ============================================================
__global__ __launch_bounds__(256) void hmulr_kernel(const float* __restrict__ h,
                                                    const unsigned short* __restrict__ r,
                                                    unsigned short* __restrict__ hr) {
    size_t t = (size_t)blockIdx.x * 256 + threadIdx.x;
    hr[t] = f2bf(h[t] * bf2f(r[t]));
}

// ============================================================
// out = x + u*(h - x)   (u already sigmoid'd, bf16)
// ============================================================
__global__ __launch_bounds__(256) void combine_kernel(const float* __restrict__ x,
                                                      const unsigned short* __restrict__ u,
                                                      const float* __restrict__ h,
                                                      float* __restrict__ out) {
    size_t t = (size_t)blockIdx.x * 256 + threadIdx.x;
    float uu = bf2f(u[t]);
    out[t] = x[t] + uu * (h[t] - x[t]);
}

// ============================================================
// host-side orchestration
// ============================================================
extern "C" void kernel_launch(void* const* d_in, const int* in_sizes, int n_in,
                              void* d_out, int out_size, void* d_ws, size_t ws_size,
                              hipStream_t stream) {
    (void)in_sizes; (void)n_in; (void)out_size; (void)ws_size;
    const float* x        = (const float*)d_in[0];
    const float* n1w      = (const float*)d_in[1];
    const float* n1b      = (const float*)d_in[2];
    const float* ema_del  = (const float*)d_in[3];
    const float* ema_alp  = (const float*)d_in[4];
    const float* ema_bet  = (const float*)d_in[5];
    const float* ema_gam  = (const float*)d_in[6];
    const float* ema_om   = (const float*)d_in[7];
    const float* vproj_w  = (const float*)d_in[8];
    const float* vproj_b  = (const float*)d_in[9];
    const float* mx_w     = (const float*)d_in[10];
    const float* mx_bias  = (const float*)d_in[11];
    const float* h_w      = (const float*)d_in[12];
    const float* h_b      = (const float*)d_in[13];
    const float* att_g    = (const float*)d_in[14];
    const float* att_b    = (const float*)d_in[15];
    const float* rel_bias = (const float*)d_in[16];
    const float* n2w      = (const float*)d_in[17];
    const float* n2b      = (const float*)d_in[18];
    const float* ff_w1    = (const float*)d_in[19];
    const float* ff_b1    = (const float*)d_in[20];
    const float* ff_w2    = (const float*)d_in[21];
    const float* ff_b2    = (const float*)d_in[22];

    char* ws = (char*)d_ws;
    size_t off = 0;
    auto alloc = [&](size_t bytes) -> char* {
        size_t a = (off + 255) & ~(size_t)255;
        off = a + bytes;
        return ws + a;
    };
    typedef unsigned short u16;
    float* xn_f   = (float*)alloc((size_t)Mt * Dc * 4);
    u16*   xn_b   = (u16*)  alloc((size_t)Mt * Dc * 2);
    u16*   mx_b16 = (u16*)  alloc((size_t)Mt * Dc * 2);
    u16*   vprojT = (u16*)  alloc((size_t)Dc * Vc * 2);
    u16*   mxwT   = (u16*)  alloc((size_t)Dc * COLS * 2);
    u16*   hwT    = (u16*)  alloc((size_t)Vc * Dc * 2);
    u16*   ff1T   = (u16*)  alloc((size_t)Dc * 2 * Dc * 2);
    u16*   ff2T   = (u16*)  alloc((size_t)2 * Dc * Dc * 2);
    u16*   v_t    = (u16*)  alloc((size_t)Vc * Mt * 2);      // [V][Mt] transposed v
    u16*   u_b    = (u16*)  alloc((size_t)Mt * Dc * 2);
    u16*   z_b    = (u16*)  alloc((size_t)Mt * Zc * 2);
    u16*   r_b    = (u16*)  alloc((size_t)Mt * Vc * 2);
    float* hx_f   = (float*)alloc((size_t)Mt * Dc * 4);
    u16*   qh_b   = (u16*)  alloc((size_t)Mt * Zc * 2);
    u16*   kh_b   = (u16*)  alloc((size_t)Mt * Zc * 2);
    float* sc_f   = (float*)alloc((size_t)Lc * Lc * 4);      // per-batch scores
    u16*   at_b   = (u16*)  alloc((size_t)Lc * Lc * 2);      // per-batch attn
    float* hb_f   = (float*)alloc((size_t)Lc * Vc * 4);      // per-batch h
    u16*   hr_b   = (u16*)  alloc((size_t)Mt * Vc * 2);
    float* hout_f = (float*)alloc((size_t)Mt * Dc * 4);
    float* out_f  = (float*)alloc((size_t)Mt * Dc * 4);
    u16*   yn_b   = (u16*)  alloc((size_t)Mt * Dc * 2);
    u16*   ff1_b  = (u16*)  alloc((size_t)Mt * 2 * Dc * 2);

    // 1) LN1 -> xn (f32 for EMA, bf16 for GEMM)
    ln_kernel<<<Mt, 256, 0, stream>>>(x, n1w, n1b, xn_f, xn_b);

    // 2) EMA recurrence + silu gate -> mx bf16
    ema_mx_kernel<<<(Bc * Dc) / 256, 256, 0, stream>>>(xn_f, ema_del, ema_alp, ema_bet,
                                                       ema_gam, ema_om, mx_b16);

    // 3) weight transposes to bf16 [N][K]
    wtrans_kernel<<<(Dc * Vc) / 256, 256, 0, stream>>>(vproj_w, vprojT, Dc, Vc);
    wtrans_kernel<<<(Dc * COLS) / 256, 256, 0, stream>>>(mx_w, mxwT, Dc, COLS);
    wtrans_kernel<<<(Vc * Dc) / 256, 256, 0, stream>>>(h_w, hwT, Vc, Dc);
    wtrans_kernel<<<(Dc * 2 * Dc) / 256, 256, 0, stream>>>(ff_w1, ff1T, Dc, 2 * Dc);
    wtrans_kernel<<<(2 * Dc * Dc) / 256, 256, 0, stream>>>(ff_w2, ff2T, 2 * Dc, Dc);

    // 4) v = silu(xn @ vproj + b), stored TRANSPOSED [V][Mt] for the attn.v GEMM
    gemm_wmma<1, 2><<<dim3(Vc / 128, Mt / 64), 256, 0, stream>>>(
        xn_b, vprojT, vproj_b, nullptr, nullptr, v_t, Dc, Dc, Dc, Mt);

    // 5) base = mx @ mx_w + mx_b, split into 4 column chunks with fused epilogues
    gemm_wmma<2, 1><<<dim3(Dc / 128, Mt / 64), 256, 0, stream>>>(            // u = sigmoid
        mx_b16, mxwT, mx_bias, nullptr, nullptr, u_b, Dc, Dc, Dc, Dc);
    gemm_wmma<1, 1><<<dim3(Zc / 128, Mt / 64), 256, 0, stream>>>(            // z = silu
        mx_b16, mxwT + (size_t)Dc * Dc, mx_bias + Dc, nullptr, nullptr, z_b, Dc, Dc, Dc, Zc);
    gemm_wmma<1, 1><<<dim3(Vc / 128, Mt / 64), 256, 0, stream>>>(            // r = silu
        mx_b16, mxwT + (size_t)(Dc + Zc) * Dc, mx_bias + Dc + Zc, nullptr, nullptr, r_b,
        Dc, Dc, Dc, Vc);
    gemm_wmma<0, 0><<<dim3(Dc / 128, Mt / 64), 256, 0, stream>>>(            // hx raw f32
        mx_b16, mxwT + (size_t)(Dc + Zc + Vc) * Dc, mx_bias + Dc + Zc + Vc, nullptr, hx_f,
        nullptr, Dc, Dc, Dc, Dc);

    // 6) qh/kh from z
    qhkh_kernel<<<(Mt * Zc) / 256, 256, 0, stream>>>(z_b, att_g, att_b, qh_b, kh_b);

    // 7) attention, per batch (score tile = 16MB, resident in L2)
    for (int b = 0; b < Bc; b++) {
        gemm_wmma<0, 0><<<dim3(Lc / 128, Lc / 64), 256, 0, stream>>>(        // qk^T
            qh_b + (size_t)b * Lc * Zc, kh_b + (size_t)b * Lc * Zc, nullptr, nullptr,
            sc_f, nullptr, Zc, Zc, Zc, Lc);
        softmax_kernel<<<Lc, 256, 0, stream>>>(sc_f, rel_bias, at_b);        // +bias, softmax
        gemm_wmma<0, 0><<<dim3(Vc / 128, Lc / 64), 256, 0, stream>>>(        // attn @ v
            at_b, v_t + (size_t)b * Lc, nullptr, nullptr, hb_f, nullptr, Lc, Lc, Mt, Vc);
        hmulr_kernel<<<(Lc * Vc) / 256, 256, 0, stream>>>(                   // h * r -> bf16
            hb_f, r_b + (size_t)b * Lc * Vc, hr_b + (size_t)b * Lc * Vc);
    }

    // 8) h = silu(hx + (h*r) @ h_w + h_b)   (hx fused as addend)
    gemm_wmma<1, 0><<<dim3(Dc / 128, Mt / 64), 256, 0, stream>>>(
        hr_b, hwT, h_b, hx_f, hout_f, nullptr, Vc, Vc, Vc, Dc);

    // 9) out = x + u*(h - x)
    combine_kernel<<<(Mt * Dc) / 256, 256, 0, stream>>>(x, u_b, hout_f, out_f);

    // 10) FFN: y = silu(LN2(out) @ ff_w1 + b1) @ ff_w2 + b2 ; d_out = out + y
    ln_kernel<<<Mt, 256, 0, stream>>>(out_f, n2w, n2b, nullptr, yn_b);
    gemm_wmma<1, 1><<<dim3((2 * Dc) / 128, Mt / 64), 256, 0, stream>>>(
        yn_b, ff1T, ff_b1, nullptr, nullptr, ff1_b, Dc, Dc, Dc, 2 * Dc);
    gemm_wmma<0, 0><<<dim3(Dc / 128, Mt / 64), 256, 0, stream>>>(
        ff1_b, ff2T, ff_b2, out_f, (float*)d_out, nullptr, 2 * Dc, 2 * Dc, 2 * Dc, Dc);
}
